// MLP_52106543235557
// MI455X (gfx1250) — matrix operand order repaired and checked by reading, not compiled, not run on hardware
//
#include <hip/hip_runtime.h>
#include <hip/hip_bf16.h>

typedef __attribute__((ext_vector_type(16))) _Float16 v16h;
typedef __attribute__((ext_vector_type(8)))  _Float16 v8h;
typedef __attribute__((ext_vector_type(4)))  _Float16 v4h;
typedef __attribute__((ext_vector_type(8)))  float    v8f;
typedef __attribute__((ext_vector_type(4)))  float    v4f;
typedef __attribute__((ext_vector_type(4)))  int      v4i;

#define DIM     1024
#define HIDDEN  3072
#define TOKENS  8192
#define QSCALE  (1.0f / 127.0f)

// Block computes BM x BN; 8 waves (4 in M, 2 in N), each wave 32x32.
#define BM 128
#define BN 64
#define BK 64            // two WMMA K-steps per staged tile -> half the barriers
#define LDSP (BK + 8)    // 72 halves = 144B pitch (16B aligned, bank-spread)

// LDS byte offset of a generic pointer into shared memory (for async-to-LDS).
__device__ __forceinline__ unsigned lds_off(const void* p) {
    return (unsigned)(unsigned long long)
        (__attribute__((address_space(3))) const char*)p;
}

// Per-lane async 16B copy global -> LDS (tracked by ASYNCcnt).
__device__ __forceinline__ void async_copy_b128(unsigned dst_lds, const void* src) {
    asm volatile("global_load_async_to_lds_b128 %0, %1, off"
                 :: "v"(dst_lds), "v"(src) : "memory");
}
__device__ __forceinline__ void wait_asynccnt0() {
    asm volatile("s_wait_asynccnt 0x0" ::: "memory");
}

// ---------------------------------------------------------------------------
// Kernel 1: h = silu(s * x@Wg^T) * (s * x@Wu^T), stored f16 into d_ws.
// Register-prefetched staging: load tile k+1 to VGPRs while WMMAs run on tile k.
// ---------------------------------------------------------------------------
__global__ __launch_bounds__(256)
void mlp_gateup_silu_kernel(const float* __restrict__ x,
                            const int*   __restrict__ wg,
                            const int*   __restrict__ wu,
                            _Float16*    __restrict__ hbuf)
{
    __shared__ _Float16 Ah[BM][LDSP];
    __shared__ _Float16 Bg[BN][LDSP];
    __shared__ _Float16 Bu[BN][LDSP];

    const int tid   = threadIdx.x;
    const int lane  = tid & 31;
    const int wave  = tid >> 5;
    const int wm    = wave & 3;
    const int wn    = wave >> 2;
    const int lhalf = lane >> 4;
    const int lmod  = lane & 15;

    const int m0 = blockIdx.x * BM;
    const int n0 = blockIdx.y * BN;

    v8f accg[2][2] = {};
    v8f accu[2][2] = {};

    v4f ar[8];            // A prefetch regs: 128x64 f32 = 2048 float4 / 256 thr
    v4i gr[4], ur[4];     // B prefetch regs: 64x64 int32 = 1024 int4 / 256 thr

    // ---- prologue: issue loads for k0 = 0 ----
    #pragma unroll
    for (int t = 0; t < 8; ++t) {
        const int s = tid + t * 256;
        ar[t] = *(const v4f*)&x[(size_t)(m0 + (s >> 4)) * DIM + (s & 15) * 4];
    }
    #pragma unroll
    for (int t = 0; t < 4; ++t) {
        const int s = tid + t * 256;
        const size_t off = (size_t)(n0 + (s >> 4)) * DIM + (s & 15) * 4;
        gr[t] = *(const v4i*)&wg[off];
        ur[t] = *(const v4i*)&wu[off];
    }

    for (int k0 = 0; k0 < DIM; k0 += BK) {
        // ---- drain prefetch regs into LDS (single wait point) ----
        #pragma unroll
        for (int t = 0; t < 8; ++t) {
            const int s = tid + t * 256;
            const int row = s >> 4, c4 = (s & 15) * 4;
            v4h hv;
            hv[0] = (_Float16)ar[t][0]; hv[1] = (_Float16)ar[t][1];
            hv[2] = (_Float16)ar[t][2]; hv[3] = (_Float16)ar[t][3];
            *(v4h*)&Ah[row][c4] = hv;
        }
        #pragma unroll
        for (int t = 0; t < 4; ++t) {
            const int s = tid + t * 256;
            const int row = s >> 4, c4 = (s & 15) * 4;
            v4h gv, uv;
            gv[0] = (_Float16)gr[t][0]; gv[1] = (_Float16)gr[t][1];
            gv[2] = (_Float16)gr[t][2]; gv[3] = (_Float16)gr[t][3];
            uv[0] = (_Float16)ur[t][0]; uv[1] = (_Float16)ur[t][1];
            uv[2] = (_Float16)ur[t][2]; uv[3] = (_Float16)ur[t][3];
            *(v4h*)&Bg[row][c4] = gv;
            *(v4h*)&Bu[row][c4] = uv;
        }
        __syncthreads();

        // ---- issue next tile's global loads (overlap with WMMAs below) ----
        const int kn = k0 + BK;
        if (kn < DIM) {
            #pragma unroll
            for (int t = 0; t < 8; ++t) {
                const int s = tid + t * 256;
                ar[t] = *(const v4f*)&x[(size_t)(m0 + (s >> 4)) * DIM + kn + (s & 15) * 4];
            }
            #pragma unroll
            for (int t = 0; t < 4; ++t) {
                const int s = tid + t * 256;
                const size_t off = (size_t)(n0 + (s >> 4)) * DIM + kn + (s & 15) * 4;
                gr[t] = *(const v4i*)&wg[off];
                ur[t] = *(const v4i*)&wu[off];
            }
        }

        // ---- 2 WMMA K-substeps from LDS ----
        #pragma unroll
        for (int kk = 0; kk < BK; kk += 32) {
            v16h afrag[2], bgfrag[2], bufrag[2];
            #pragma unroll
            for (int i = 0; i < 2; ++i) {
                const int row = wm * 32 + i * 16 + lmod;
                v8h lo = *(const v8h*)&Ah[row][kk + lhalf * 16];
                v8h hi = *(const v8h*)&Ah[row][kk + lhalf * 16 + 8];
                afrag[i] = __builtin_shufflevector(lo, hi,
                    0, 1, 2, 3, 4, 5, 6, 7, 8, 9, 10, 11, 12, 13, 14, 15);
            }
            #pragma unroll
            for (int j = 0; j < 2; ++j) {
                const int row = wn * 32 + j * 16 + lmod;
                bgfrag[j] = *(const v16h*)&Bg[row][kk + lhalf * 16];
                bufrag[j] = *(const v16h*)&Bu[row][kk + lhalf * 16];
            }
            #pragma unroll
            for (int i = 0; i < 2; ++i) {
                #pragma unroll
                for (int j = 0; j < 2; ++j) {
                    accg[i][j] = __builtin_amdgcn_wmma_f32_16x16x32_f16(
                        false, afrag[i], false, bgfrag[j], (short)0, accg[i][j],
                        false, false);
                    accu[i][j] = __builtin_amdgcn_wmma_f32_16x16x32_f16(
                        false, afrag[i], false, bufrag[j], (short)0, accu[i][j],
                        false, false);
                }
            }
        }
        __syncthreads();
    }

    // ---- epilogue: dequant-scale, SiLU * up, store f16 ----
    #pragma unroll
    for (int i = 0; i < 2; ++i) {
        #pragma unroll
        for (int j = 0; j < 2; ++j) {
            const int n = n0 + wn * 32 + j * 16 + lmod;
            #pragma unroll
            for (int r = 0; r < 8; ++r) {
                const int m = m0 + wm * 32 + i * 16 + lhalf * 8 + r;
                float g = accg[i][j][r] * QSCALE;
                float u = accu[i][j][r] * QSCALE;
                float sig = __frcp_rn(1.0f + __expf(-g));
                hbuf[(size_t)m * HIDDEN + n] = (_Float16)(g * sig * u);
            }
        }
    }
}

// ---------------------------------------------------------------------------
// Kernel 2: out = s * (h @ Wd^T), f32 output.
// A-tile (pure f16 copy) staged via GLOBAL_LOAD_ASYNC_TO_LDS_B128 (ASYNCcnt);
// B-tile (int->f16) register-prefetched.
// ---------------------------------------------------------------------------
__global__ __launch_bounds__(256)
void mlp_down_kernel(const _Float16* __restrict__ h,
                     const int*      __restrict__ wd,
                     float*          __restrict__ out)
{
    __shared__ _Float16 Ah[BM][LDSP];
    __shared__ _Float16 Bd[BN][LDSP];

    const int tid   = threadIdx.x;
    const int lane  = tid & 31;
    const int wave  = tid >> 5;
    const int wm    = wave & 3;
    const int wn    = wave >> 2;
    const int lhalf = lane >> 4;
    const int lmod  = lane & 15;

    const int m0 = blockIdx.x * BM;
    const int n0 = blockIdx.y * BN;

    v8f acc[2][2] = {};
    v4i br[4];     // B prefetch regs: 64x64 int32 = 1024 int4 / 256 thr

    // ---- prologue: async A copy + B loads for k0 = 0 ----
    #pragma unroll
    for (int t = 0; t < 4; ++t) {             // 128x64 f16 = 1024 x 16B chunks
        const int s = tid + t * 256;
        const int row = s >> 3, c8 = (s & 7) * 8;
        async_copy_b128(lds_off(&Ah[row][c8]),
                        &h[(size_t)(m0 + row) * HIDDEN + c8]);
    }
    #pragma unroll
    for (int t = 0; t < 4; ++t) {
        const int s = tid + t * 256;
        br[t] = *(const v4i*)&wd[(size_t)(n0 + (s >> 4)) * HIDDEN + (s & 15) * 4];
    }

    for (int k0 = 0; k0 < HIDDEN; k0 += BK) {
        // ---- drain B regs into LDS; fence this wave's async A copies ----
        #pragma unroll
        for (int t = 0; t < 4; ++t) {
            const int s = tid + t * 256;
            const int row = s >> 4, c4 = (s & 15) * 4;
            v4h wv;
            wv[0] = (_Float16)br[t][0]; wv[1] = (_Float16)br[t][1];
            wv[2] = (_Float16)br[t][2]; wv[3] = (_Float16)br[t][3];
            *(v4h*)&Bd[row][c4] = wv;
        }
        wait_asynccnt0();
        __syncthreads();

        // ---- 2 WMMA K-substeps from LDS ----
        #pragma unroll
        for (int kk = 0; kk < BK; kk += 32) {
            v16h afrag[2], bfrag[2];
            #pragma unroll
            for (int i = 0; i < 2; ++i) {
                const int row = wm * 32 + i * 16 + lmod;
                v8h lo = *(const v8h*)&Ah[row][kk + lhalf * 16];
                v8h hi = *(const v8h*)&Ah[row][kk + lhalf * 16 + 8];
                afrag[i] = __builtin_shufflevector(lo, hi,
                    0, 1, 2, 3, 4, 5, 6, 7, 8, 9, 10, 11, 12, 13, 14, 15);
            }
            #pragma unroll
            for (int j = 0; j < 2; ++j) {
                const int row = wn * 32 + j * 16 + lmod;
                bfrag[j] = *(const v16h*)&Bd[row][kk + lhalf * 16];
            }
            #pragma unroll
            for (int i = 0; i < 2; ++i) {
                #pragma unroll
                for (int j = 0; j < 2; ++j) {
                    acc[i][j] = __builtin_amdgcn_wmma_f32_16x16x32_f16(
                        false, afrag[i], false, bfrag[j], (short)0, acc[i][j],
                        false, false);
                }
            }
        }
        __syncthreads();

        // ---- issue next tile: async A copy + B global loads ----
        const int kn = k0 + BK;
        if (kn < HIDDEN) {
            #pragma unroll
            for (int t = 0; t < 4; ++t) {
                const int s = tid + t * 256;
                const int row = s >> 3, c8 = (s & 7) * 8;
                async_copy_b128(lds_off(&Ah[row][c8]),
                                &h[(size_t)(m0 + row) * HIDDEN + kn + c8]);
            }
            #pragma unroll
            for (int t = 0; t < 4; ++t) {
                const int s = tid + t * 256;
                br[t] = *(const v4i*)&wd[(size_t)(n0 + (s >> 4)) * HIDDEN + kn + (s & 15) * 4];
            }
        }
    }

    #pragma unroll
    for (int i = 0; i < 2; ++i) {
        #pragma unroll
        for (int j = 0; j < 2; ++j) {
            const int n = n0 + wn * 32 + j * 16 + lmod;
            #pragma unroll
            for (int r = 0; r < 8; ++r) {
                const int m = m0 + wm * 32 + i * 16 + lhalf * 8 + r;
                out[(size_t)m * DIM + n] = acc[i][j][r] * QSCALE;
            }
        }
    }
}

// ---------------------------------------------------------------------------
extern "C" void kernel_launch(void* const* d_in, const int* in_sizes, int n_in,
                              void* d_out, int out_size, void* d_ws, size_t ws_size,
                              hipStream_t stream) {
    const float* x   = (const float*)d_in[0];
    const int*   wg  = (const int*)d_in[1];
    const int*   wu  = (const int*)d_in[2];
    const int*   wd  = (const int*)d_in[3];
    float*       out = (float*)d_out;
    _Float16*    hbuf = (_Float16*)d_ws;   // TOKENS*HIDDEN f16 = 48 MB scratch

    dim3 blk(256);
    mlp_gateup_silu_kernel<<<dim3(TOKENS / BM, HIDDEN / BN), blk, 0, stream>>>(
        x, wg, wu, hbuf);
    mlp_down_kernel<<<dim3(TOKENS / BM, DIM / BN), blk, 0, stream>>>(
        hbuf, wd, out);
}
